// NestedHydra_70746701300043
// MI455X (gfx1250) — matrix operand order repaired and checked by reading, not compiled, not run on hardware
//
#include <hip/hip_runtime.h>
#include <hip/hip_bf16.h>
#include <math.h>

// ---------------------------------------------------------------------------
// Bidirectional Mamba-2 (Hydra) block for MI455X (gfx1250, wave32, WMMA).
// All GEMM-shaped work funnels through v_wmma_f32_16x16x32_f16 with exact
// tile shapes (no guards): 64x64 block tile, 8 waves, 2 WMMA tiles/wave,
// b128 LDS fragment loads matching the ISA wave32 A/B layouts.
// ---------------------------------------------------------------------------

#define DIMD    768
#define INNER   1536
#define NSTATE  64
#define NHEADS  8
#define HEADDIM 192
#define KW      7
#define CHUNK   256
#define LSEQ    1024
#define CONV_C  1664          // INNER + 2*NSTATE
#define KCONV   (CONV_C*KW)   // 11648
#define NCHUNK  4
#define DTW     64            // padded width of dt projection (16 real cols)

typedef _Float16 half_t;
typedef half_t v16h __attribute__((ext_vector_type(16)));
typedef half_t v8h  __attribute__((ext_vector_type(8)));
typedef half_t v4h  __attribute__((ext_vector_type(4)));
typedef float  v8f  __attribute__((ext_vector_type(8)));

// ---------------------------------------------------------------------------
// Batched tiled GEMM:  C[M,N](f32) (+)= A[M,K](f16,lda) @ B[K,N](f16,ldb)
// Contract: M%64==0, N%64==0, K%32==0 (caller guarantees; no guards).
// Block: 256 threads = 8 waves; tile 64(M) x 64(N); K step 32.
// Wave grid 2(M)x4(N); each wave computes two 16x16 WMMA tiles sharing B.
// ---------------------------------------------------------------------------
__global__ void __launch_bounds__(256)
gemm_wmma(const half_t* __restrict__ A, long long lda, long long sA,
          const half_t* __restrict__ B, long long ldb, long long sB,
          float* __restrict__ C, long long ldc, long long sC,
          int M, int N, int K, int acc)
{
    __shared__ half_t As[64][40];    // [m][k], 80B row stride (16B aligned)
    __shared__ half_t BsT[64][40];   // [n][k] transposed tile

    A += (long long)blockIdx.z * sA + (long long)(blockIdx.y * 64) * lda;
    B += (long long)blockIdx.z * sB + (long long)(blockIdx.x * 64);
    C += (long long)blockIdx.z * sC + (long long)(blockIdx.y * 64) * ldc
                                    + (long long)(blockIdx.x * 64);

    const int tid   = threadIdx.x;
    const int wave  = tid >> 5;
    const int lane  = tid & 31;
    const int wm    = wave >> 2;     // 0..1  (two 32-row halves)
    const int wn    = wave & 3;      // 0..3  (four 16-col strips)
    const int row16 = lane & 15;
    const int hi    = lane >> 4;

    // staging coordinates
    const int ar = tid >> 2;         // A row 0..63
    const int ac = (tid & 3) * 8;    // A k-offset 0,8,16,24
    const int kr = tid >> 3;         // B k-row 0..31
    const int c8 = (tid & 7) * 8;    // B col-offset

    v8f c0, c1;
    #pragma unroll
    for (int r = 0; r < 8; ++r) { c0[r] = 0.f; c1[r] = 0.f; }
    const int ccol = wn*16 + row16;
    if (acc) {
        #pragma unroll
        for (int r = 0; r < 8; ++r) {
            c0[r] = C[(long long)(wm*32      + r + 8*hi)*ldc + ccol];
            c1[r] = C[(long long)(wm*32 + 16 + r + 8*hi)*ldc + ccol];
        }
    }

    for (int k0 = 0; k0 < K; k0 += 32) {
        // stage A 64x32 (one b128 per thread)
        v8h av = *(const v8h*)(A + (long long)ar*lda + k0 + ac);
        *(v8h*)&As[ar][ac] = av;
        // stage B 32x64 transposed (one b128 load, 8 b16 scatter stores)
        v8h bv = *(const v8h*)(B + (long long)(k0 + kr)*ldb + c8);
        #pragma unroll
        for (int q = 0; q < 8; ++q) BsT[c8 + q][kr] = bv[q];
        if (k0 + 32 < K && tid == 0)
            __builtin_prefetch(B + (long long)(k0 + 32)*ldb, 0, 1);
        __syncthreads();

        // A fragment (16-bit A 16x32 layout) = two contiguous v8h per lane
        const int ra0 = wm*32 + row16;
        const int ra1 = ra0 + 16;
        v8h a0lo = *(const v8h*)&As[ra0][8*hi];
        v8h a0hi = *(const v8h*)&As[ra0][16 + 8*hi];
        v8h a1lo = *(const v8h*)&As[ra1][8*hi];
        v8h a1hi = *(const v8h*)&As[ra1][16 + 8*hi];
        // B fragment (same k order as A: 8*hi and 16+8*hi) = two contiguous v8h
        v8h b0 = *(const v8h*)&BsT[wn*16 + row16][8*hi];
        v8h b1 = *(const v8h*)&BsT[wn*16 + row16][16 + 8*hi];

        v16h af0 = __builtin_shufflevector(a0lo, a0hi, 0,1,2,3,4,5,6,7,8,9,10,11,12,13,14,15);
        v16h af1 = __builtin_shufflevector(a1lo, a1hi, 0,1,2,3,4,5,6,7,8,9,10,11,12,13,14,15);
        v16h bf  = __builtin_shufflevector(b0,   b1,   0,1,2,3,4,5,6,7,8,9,10,11,12,13,14,15);

        c0 = __builtin_amdgcn_wmma_f32_16x16x32_f16(false, af0, false, bf, (short)0, c0, false, false);
        c1 = __builtin_amdgcn_wmma_f32_16x16x32_f16(false, af1, false, bf, (short)0, c1, false, false);
        __syncthreads();
    }

    #pragma unroll
    for (int r = 0; r < 8; ++r) {
        C[(long long)(wm*32      + r + 8*hi)*ldc + ccol] = c0[r];
        C[(long long)(wm*32 + 16 + r + 8*hi)*ldc + ccol] = c1[r];
    }
}

// ---------------------------------------------------------------------------
// Elementwise / pack / scan kernels
// ---------------------------------------------------------------------------
__global__ void cvt_f16v4(const float* __restrict__ in, half_t* __restrict__ out, long long n4) {
    long long i = (long long)blockIdx.x*blockDim.x + threadIdx.x;
    if (i >= n4) return;
    float4 v = ((const float4*)in)[i];
    v4h o; o[0] = (half_t)v.x; o[1] = (half_t)v.y; o[2] = (half_t)v.z; o[3] = (half_t)v.w;
    ((v4h*)out)[i] = o;
}

// WdtPad[768][64]: cols 0..15 = Wdt, rest 0
__global__ void pack_Wdt(const float* __restrict__ w, half_t* __restrict__ out) {
    int idx = blockIdx.x*blockDim.x + threadIdx.x;
    if (idx >= DIMD*DTW) return;
    int c = idx & 63, r = idx >> 6;
    out[idx] = (c < 2*NHEADS) ? (half_t)w[r*2*NHEADS + c] : (half_t)0.f;
}

// w2[(k*C+i)*C + o] = conv_w[o*C*KW + i*KW + k]  (pre-transposed conv weight)
__global__ void conv_w_xform(const float* __restrict__ w, half_t* __restrict__ w2) {
    long long idx = (long long)blockIdx.x*blockDim.x + threadIdx.x;
    long long total = (long long)KW*CONV_C*CONV_C;
    if (idx >= total) return;
    int o = (int)(idx % CONV_C);
    long long r = idx / CONV_C;
    int i = (int)(r % CONV_C);
    int k = (int)(r / CONV_C);
    w2[idx] = (half_t)w[((long long)o*CONV_C + i)*KW + k];
}

// xpad[(t+3)][c] = xBC_pre[t][c]; 3 zero rows top/bottom (rows = LSEQ+6)
__global__ void build_xpad(const float* __restrict__ xbc, half_t* __restrict__ xpad) {
    long long i4 = (long long)blockIdx.x*blockDim.x + threadIdx.x;
    long long total4 = (long long)(LSEQ+6)*CONV_C/4;
    if (i4 >= total4) return;
    long long idx = i4*4;
    int col = (int)(idx % CONV_C);
    int row = (int)(idx / CONV_C);
    int src = row - 3;
    v4h o;
    if (src >= 0 && src < LSEQ) {
        float4 v = *(const float4*)&xbc[(long long)src*CONV_C + col];
        o[0]=(half_t)v.x; o[1]=(half_t)v.y; o[2]=(half_t)v.z; o[3]=(half_t)v.w;
    } else {
        o[0]=(half_t)0.f; o[1]=(half_t)0.f; o[2]=(half_t)0.f; o[3]=(half_t)0.f;
    }
    *(v4h*)&xpad[idx] = o;
}

__global__ void bias_silu(float* __restrict__ xbc, const float* __restrict__ bias) {
    long long i4 = (long long)blockIdx.x*blockDim.x + threadIdx.x;
    if (i4 >= (long long)LSEQ*CONV_C/4) return;
    long long idx = i4*4;
    int c = (int)(idx % CONV_C);
    float4 v = *(float4*)&xbc[idx];
    float4 b = *(const float4*)&bias[c];
    v.x += b.x; v.y += b.y; v.z += b.z; v.w += b.w;
    v.x /= (1.f + expf(-v.x)); v.y /= (1.f + expf(-v.y));
    v.z /= (1.f + expf(-v.z)); v.w /= (1.f + expf(-v.w));
    *(float4*)&xbc[idx] = v;
}

// dt softplus + per-chunk inclusive cumsum of dA
__global__ void __launch_bounds__(256)
dt_scan(const float* __restrict__ dtraw, const float* __restrict__ dt_bias,
        const float* __restrict__ log_A, float* __restrict__ dt2,
        float* __restrict__ dA_cs, float* __restrict__ expcs,
        float* __restrict__ decayS, float* __restrict__ chunk_decay)
{
    int bx = blockIdx.x;                 // 64 = d*32 + h*4 + c
    int d = bx >> 5, h = (bx >> 2) & 7, c = bx & 3;
    int j = threadIdx.x;
    int t  = c*CHUNK + j;
    int tt = d ? (LSEQ-1 - t) : t;
    float raw = dtraw[(long long)tt*DTW + (d ? NHEADS + h : h)];
    float xv  = raw + dt_bias[h];
    float sp  = (xv > 20.f) ? xv : log1pf(expf(xv));
    dt2[((long long)d*LSEQ + t)*NHEADS + h] = sp;
    float da = sp * (-expf(log_A[h]));

    __shared__ float sbuf[CHUNK];
    sbuf[j] = da; __syncthreads();
    for (int off = 1; off < CHUNK; off <<= 1) {
        float v = (j >= off) ? sbuf[j-off] : 0.f;
        __syncthreads();
        sbuf[j] += v;
        __syncthreads();
    }
    float cs = sbuf[j], last = sbuf[CHUNK-1];
    long long base = ((((long long)d*NHEADS + h)*NCHUNK) + c)*CHUNK + j;
    dA_cs[base]  = cs;
    expcs[base]  = expf(cs);
    decayS[base] = expf(last - cs);
    if (j == 0) chunk_decay[((long long)d*NHEADS + h)*NCHUNK + c] = expf(last);
}

// Cc16[d][t][n] = xbc[tt][1600+n]
__global__ void pack_Cc(const float* __restrict__ xbc, half_t* __restrict__ out) {
    long long idx = (long long)blockIdx.x*blockDim.x + threadIdx.x;
    if (idx >= (1LL<<17)) return;
    int n = idx & 63, t = (idx >> 6) & 1023, d = (int)(idx >> 16);
    int tt = d ? (LSEQ-1 - t) : t;
    out[idx] = (half_t)xbc[(long long)tt*CONV_C + INNER + NSTATE + n];
}

// BcT16[d][c][n][j] = xbc[tt][1536+n]
__global__ void pack_BcT(const float* __restrict__ xbc, half_t* __restrict__ out) {
    long long idx = (long long)blockIdx.x*blockDim.x + threadIdx.x;
    if (idx >= (1LL<<17)) return;
    int j = idx & 255, n = (idx >> 8) & 63, c = (idx >> 14) & 3, d = (int)(idx >> 16);
    int t = c*CHUNK + j;
    int tt = d ? (LSEQ-1 - t) : t;
    out[idx] = (half_t)xbc[(long long)tt*CONV_C + INNER + n];
}

// xd16[d][c][h][j][p] = x[tt][h*192+p] * dt2[d][t][h]
__global__ void pack_xd(const float* __restrict__ xbc, const float* __restrict__ dt2,
                        half_t* __restrict__ out) {
    long long idx = (long long)blockIdx.x*blockDim.x + threadIdx.x;
    long long total = 2LL*NCHUNK*NHEADS*CHUNK*HEADDIM;
    if (idx >= total) return;
    int p = (int)(idx % HEADDIM);
    long long r = idx / HEADDIM;
    int j = (int)(r & 255); r >>= 8;
    int h = (int)(r & 7);   r >>= 3;
    int c = (int)(r & 3);
    int d = (int)(r >> 2);
    int t = c*CHUNK + j;
    int tt = d ? (LSEQ-1 - t) : t;
    float xv  = xbc[(long long)tt*CONV_C + h*HEADDIM + p];
    float dtv = dt2[((long long)d*LSEQ + t)*NHEADS + h];
    out[idx] = (half_t)(xv * dtv);
}

// Ah16[d][c][h][i][j] = (j<=i) ? G[d][c][i][j]*exp(cs[i]-cs[j]) : 0
__global__ void pack_Ah(const float* __restrict__ G, const float* __restrict__ cs,
                        half_t* __restrict__ out) {
    long long idx = (long long)blockIdx.x*blockDim.x + threadIdx.x;
    if (idx >= (1LL<<22)) return;
    int j = idx & 255, i = (idx >> 8) & 255;
    int h = (idx >> 16) & 7, c = (idx >> 19) & 3, d = (int)(idx >> 21);
    float v = 0.f;
    if (j <= i) {
        float g = G[((((long long)d*NCHUNK + c)*CHUNK + i) << 8) + j];
        long long cb = ((((long long)d*NHEADS + h)*NCHUNK) + c)*CHUNK;
        v = g * expf(cs[cb + i] - cs[cb + j]);
    }
    out[idx] = (half_t)v;
}

// Bdec16[d][c][h][n][j] = BcT16[d][c][n][j] * decayS[d][h][c][j]
__global__ void pack_Bdec(const half_t* __restrict__ BcT, const float* __restrict__ dec,
                          half_t* __restrict__ out) {
    long long idx = (long long)blockIdx.x*blockDim.x + threadIdx.x;
    if (idx >= (1LL<<20)) return;
    int j = idx & 255, n = (idx >> 8) & 63;
    int h = (idx >> 14) & 7, c = (idx >> 17) & 3, d = (int)(idx >> 19);
    float b = (float)BcT[((((long long)d*NCHUNK + c)*NSTATE + n) << 8) + j];
    float e = dec[(((((long long)d*NHEADS + h)*NCHUNK) + c) << 8) + j];
    out[idx] = (half_t)(b * e);
}

// CcS16[d][c][h][i][n] = Cc16[d][c*256+i][n] * expcs[d][h][c][i]
__global__ void pack_CcS(const half_t* __restrict__ Cc, const float* __restrict__ ecs,
                         half_t* __restrict__ out) {
    long long idx = (long long)blockIdx.x*blockDim.x + threadIdx.x;
    if (idx >= (1LL<<20)) return;
    int n = idx & 63, i = (idx >> 6) & 255;
    int h = (idx >> 14) & 7, c = (idx >> 17) & 3, d = (int)(idx >> 19);
    float cv = (float)Cc[(((long long)d*LSEQ + c*CHUNK + i) << 6) + n];
    float e  = ecs[(((((long long)d*NHEADS + h)*NCHUNK) + c) << 8) + i];
    out[idx] = (half_t)(cv * e);
}

// Sequential scan over chunks: prev[c] = state entering chunk c (f16 out).
__global__ void state_scan(const float* __restrict__ statesT, const float* __restrict__ cdec,
                           const float* __restrict__ init_state, half_t* __restrict__ prev16) {
    int b = blockIdx.x, d = b >> 3, h = b & 7;   // 16 blocks
    for (int e = threadIdx.x; e < NSTATE*HEADDIM; e += blockDim.x) {
        int n = e / HEADDIM, p = e % HEADDIM;
        float carry = init_state[((long long)h*HEADDIM + p)*NSTATE + n];
        for (int c = 0; c < NCHUNK; ++c) {
            long long ob = ((((long long)d*NCHUNK + c)*NHEADS + h)*NSTATE + n)*HEADDIM + p;
            prev16[ob] = (half_t)carry;
            carry = cdec[((long long)d*NHEADS + h)*NCHUNK + c] * carry + statesT[ob];
        }
    }
}

// Dterm[t][h] = sum_f x_og[t][f] * D_w[f][h] + D_b[h]
__global__ void dterm_kernel(const float* __restrict__ xbc, const float* __restrict__ Dw,
                             const float* __restrict__ Db, float* __restrict__ Dterm) {
    int idx = blockIdx.x*blockDim.x + threadIdx.x;
    if (idx >= LSEQ*NHEADS) return;
    int h = idx & 7, t = idx >> 3;
    float acc = Db[h];
    const float* xr = xbc + (long long)t*CONV_C;
    for (int f = 0; f < INNER; ++f) acc += xr[f] * Dw[(long long)f*NHEADS + h];
    Dterm[idx] = acc;
}

// ycomb = shift(y_fw) + flip(shift(y_bw)) + x_og * repeat(Dterm)
__global__ void combine_kernel(const float* __restrict__ Ydc, const float* __restrict__ xbc,
                               const float* __restrict__ Dterm, float* __restrict__ ycomb) {
    long long idx = (long long)blockIdx.x*blockDim.x + threadIdx.x;
    if (idx >= (long long)LSEQ*INNER) return;
    int f = (int)(idx % INNER), t = (int)(idx / INNER);
    int h = f / HEADDIM, p = f % HEADDIM;
    float yfw = 0.f, ybw = 0.f;
    if (t > 0) {
        int s = t - 1, c = s >> 8, i = s & 255;
        yfw = Ydc[(((long long)c*NHEADS + h)*CHUNK + i)*HEADDIM + p];
    }
    if (t < LSEQ-1) {
        int s = (LSEQ-2) - t, c = s >> 8, i = s & 255;
        ybw = Ydc[((((long long)(NCHUNK + c))*NHEADS + h)*CHUNK + i)*HEADDIM + p];
    }
    float xog = xbc[(long long)t*CONV_C + f];
    ycomb[idx] = yfw + ybw + xog * Dterm[(long long)t*NHEADS + h];
}

// RMS-style norm * norm_w * silu(z), emit f16 for final GEMM.
__global__ void __launch_bounds__(256)
norm_gate(const float* __restrict__ ycomb, const float* __restrict__ z,
          const float* __restrict__ nw, half_t* __restrict__ y16) {
    int t = blockIdx.x;
    __shared__ float red[256];
    float ss = 0.f;
    for (int f = threadIdx.x; f < INNER; f += 256) {
        float v = ycomb[(long long)t*INNER + f]; ss += v*v;
    }
    red[threadIdx.x] = ss; __syncthreads();
    for (int off = 128; off > 0; off >>= 1) {
        if (threadIdx.x < off) red[threadIdx.x] += red[threadIdx.x + off];
        __syncthreads();
    }
    float inv = rsqrtf(red[0] / (float)INNER + 1e-5f);
    for (int f = threadIdx.x; f < INNER; f += 256) {
        float zv = z[(long long)t*INNER + f];
        float v  = ycomb[(long long)t*INNER + f] * inv * nw[f] * (zv / (1.f + expf(-zv)));
        y16[(long long)t*INNER + f] = (half_t)v;
    }
}

// ---------------------------------------------------------------------------
// Host launcher
// ---------------------------------------------------------------------------
extern "C" void kernel_launch(void* const* d_in, const int* in_sizes, int n_in,
                              void* d_out, int out_size, void* d_ws, size_t ws_size,
                              hipStream_t stream) {
    const float* u       = (const float*)d_in[0];
    const float* Wz      = (const float*)d_in[2];
    const float* Wx      = (const float*)d_in[3];
    const float* WB      = (const float*)d_in[4];
    const float* WC      = (const float*)d_in[5];
    const float* Wdt     = (const float*)d_in[6];
    const float* conv_w  = (const float*)d_in[7];
    const float* conv_b  = (const float*)d_in[8];
    const float* dt_bias = (const float*)d_in[9];
    const float* log_A   = (const float*)d_in[10];
    const float* D_w     = (const float*)d_in[11];
    const float* D_b     = (const float*)d_in[12];
    const float* norm_w  = (const float*)d_in[13];
    const float* Wout    = (const float*)d_in[14];
    const float* init_st = (const float*)d_in[15];
    float* outp = (float*)d_out;

    char* p = (char*)d_ws;
    auto alloc = [&](size_t bytes) { char* r = p; p += (bytes + 255) & ~(size_t)255; return (void*)r; };

    half_t* uF16    = (half_t*)alloc(2LL*LSEQ*DIMD);
    half_t* WzF16   = (half_t*)alloc(2LL*DIMD*INNER);
    half_t* WxF16   = (half_t*)alloc(2LL*DIMD*INNER);
    half_t* WBF16   = (half_t*)alloc(2LL*DIMD*NSTATE);
    half_t* WCF16   = (half_t*)alloc(2LL*DIMD*NSTATE);
    half_t* WdtF16  = (half_t*)alloc(2LL*DIMD*DTW);
    half_t* WoutF16 = (half_t*)alloc(2LL*INNER*DIMD);
    half_t* w2F16   = (half_t*)alloc(2LL*KCONV*CONV_C);
    float*  zbuf    = (float*) alloc(4LL*LSEQ*INNER);
    float*  xBCpre  = (float*) alloc(4LL*LSEQ*CONV_C);
    float*  dtraw   = (float*) alloc(4LL*LSEQ*DTW);
    half_t* xpad    = (half_t*)alloc(2LL*(LSEQ+6)*CONV_C);
    float*  xBCpost = (float*) alloc(4LL*LSEQ*CONV_C);
    float*  dt2     = (float*) alloc(4LL*2*LSEQ*NHEADS);
    float*  dA_cs   = (float*) alloc(4LL*2*NHEADS*NCHUNK*CHUNK);
    float*  expcs   = (float*) alloc(4LL*2*NHEADS*NCHUNK*CHUNK);
    float*  decayS  = (float*) alloc(4LL*2*NHEADS*NCHUNK*CHUNK);
    float*  cdec    = (float*) alloc(4LL*2*NHEADS*NCHUNK);
    half_t* Cc16    = (half_t*)alloc(2LL*2*LSEQ*NSTATE);
    half_t* BcT16   = (half_t*)alloc(2LL*2*NCHUNK*NSTATE*CHUNK);
    half_t* xd16    = (half_t*)alloc(2LL*2*NCHUNK*NHEADS*CHUNK*HEADDIM);
    float*  Gbuf    = (float*) alloc(4LL*8*CHUNK*CHUNK);
    half_t* Ah16    = (half_t*)alloc(2LL*64*CHUNK*CHUNK);
    half_t* Bdec16  = (half_t*)alloc(2LL*64*NSTATE*CHUNK);
    half_t* CcS16   = (half_t*)alloc(2LL*64*CHUNK*NSTATE);
    float*  statesT = (float*) alloc(4LL*64*NSTATE*HEADDIM);
    half_t* prev16  = (half_t*)alloc(2LL*64*NSTATE*HEADDIM);
    float*  Ydc     = (float*) alloc(4LL*64*CHUNK*HEADDIM);
    float*  Dterm   = (float*) alloc(4LL*LSEQ*NHEADS);
    float*  ycomb   = (float*) alloc(4LL*LSEQ*INNER);
    half_t* y16     = (half_t*)alloc(2LL*LSEQ*INNER);

    auto cvt = [&](const float* src, half_t* dst, long long n) {
        long long n4 = n / 4;
        cvt_f16v4<<<dim3((unsigned)((n4 + 255) / 256)), 256, 0, stream>>>(src, dst, n4);
    };
    cvt(u,    uF16,    (long long)LSEQ*DIMD);
    cvt(Wz,   WzF16,   (long long)DIMD*INNER);
    cvt(Wx,   WxF16,   (long long)DIMD*INNER);
    cvt(WB,   WBF16,   (long long)DIMD*NSTATE);
    cvt(WC,   WCF16,   (long long)DIMD*NSTATE);
    cvt(Wout, WoutF16, (long long)INNER*DIMD);
    pack_Wdt<<<dim3((DIMD*DTW + 255) / 256), 256, 0, stream>>>(Wdt, WdtF16);

    {
        long long tot = (long long)KW*CONV_C*CONV_C;
        conv_w_xform<<<dim3((unsigned)((tot + 255) / 256)), 256, 0, stream>>>(conv_w, w2F16);
    }

    auto gemm = [&](const half_t* A, long long lda, long long sA,
                    const half_t* B, long long ldb, long long sB,
                    float* C, long long ldc, long long sC,
                    int M, int N, int K, int acc, int batch) {
        dim3 grid(N / 64, M / 64, batch);
        gemm_wmma<<<grid, 256, 0, stream>>>(A, lda, sA, B, ldb, sB, C, ldc, sC, M, N, K, acc);
    };

    // projections
    gemm(uF16, DIMD, 0, WzF16,  INNER,  0, zbuf,                INNER,  0, LSEQ, INNER,  DIMD, 0, 1);
    gemm(uF16, DIMD, 0, WxF16,  INNER,  0, xBCpre,              CONV_C, 0, LSEQ, INNER,  DIMD, 0, 1);
    gemm(uF16, DIMD, 0, WBF16,  NSTATE, 0, xBCpre+INNER,        CONV_C, 0, LSEQ, NSTATE, DIMD, 0, 1);
    gemm(uF16, DIMD, 0, WCF16,  NSTATE, 0, xBCpre+INNER+NSTATE, CONV_C, 0, LSEQ, NSTATE, DIMD, 0, 1);
    gemm(uF16, DIMD, 0, WdtF16, DTW,    0, dtraw,               DTW,    0, LSEQ, DTW,    DIMD, 0, 1);

    // conv as implicit GEMM: A = padded input with lda=CONV_C, K = CONV_C*KW
    {
        long long tot4 = (long long)(LSEQ+6)*CONV_C/4;
        build_xpad<<<dim3((unsigned)((tot4 + 255) / 256)), 256, 0, stream>>>(xBCpre, xpad);
    }
    gemm(xpad, CONV_C, 0, w2F16, CONV_C, 0, xBCpost, CONV_C, 0, LSEQ, CONV_C, KCONV, 0, 1);
    {
        long long tot4 = (long long)LSEQ*CONV_C/4;
        bias_silu<<<dim3((unsigned)((tot4 + 255) / 256)), 256, 0, stream>>>(xBCpost, conv_b);
    }

    // dt softplus + chunk cumsums
    dt_scan<<<dim3(64), 256, 0, stream>>>(dtraw, dt_bias, log_A, dt2, dA_cs, expcs, decayS, cdec);

    // SSD operand packs
    pack_Cc <<<dim3(512),  256, 0, stream>>>(xBCpost, Cc16);
    pack_BcT<<<dim3(512),  256, 0, stream>>>(xBCpost, BcT16);
    pack_xd <<<dim3((unsigned)((2LL*NCHUNK*NHEADS*CHUNK*HEADDIM + 255) / 256)), 256, 0, stream>>>(xBCpost, dt2, xd16);

    // G = Cc @ Bc^T   (batch 8 over d,c)
    gemm(Cc16, NSTATE, 256LL*NSTATE, BcT16, CHUNK, 64LL*256, Gbuf, CHUNK, 256LL*256,
         CHUNK, CHUNK, NSTATE, 0, 8);
    pack_Ah<<<dim3((unsigned)((1LL<<22) / 256)), 256, 0, stream>>>(Gbuf, dA_cs, Ah16);

    // Y_diag = (G*L) @ xd   (batch 64 over d,c,h)
    gemm(Ah16, CHUNK, 256LL*256, xd16, HEADDIM, 256LL*HEADDIM, Ydc, HEADDIM, 256LL*HEADDIM,
         CHUNK, HEADDIM, CHUNK, 0, 64);

    // statesT = (Bc*decay)^T @ xd
    pack_Bdec<<<dim3((unsigned)((1LL<<20) / 256)), 256, 0, stream>>>(BcT16, decayS, Bdec16);
    gemm(Bdec16, CHUNK, 64LL*256, xd16, HEADDIM, 256LL*HEADDIM, statesT, HEADDIM, 64LL*HEADDIM,
         NSTATE, HEADDIM, CHUNK, 0, 64);

    // inter-chunk recurrence
    state_scan<<<dim3(16), 256, 0, stream>>>(statesT, cdec, init_st, prev16);

    // Y_off += (Cc*exp(cs)) @ prev^T
    pack_CcS<<<dim3((unsigned)((1LL<<20) / 256)), 256, 0, stream>>>(Cc16, expcs, CcS16);
    gemm(CcS16, NSTATE, 256LL*NSTATE, prev16, HEADDIM, 64LL*HEADDIM, Ydc, HEADDIM, 256LL*HEADDIM,
         CHUNK, HEADDIM, NSTATE, 1, 64);

    // D-term, combine, norm+gate
    dterm_kernel<<<dim3(32), 256, 0, stream>>>(xBCpost, D_w, D_b, Dterm);
    combine_kernel<<<dim3((unsigned)((long long)LSEQ*INNER / 256)), 256, 0, stream>>>(Ydc, xBCpost, Dterm, ycomb);
    norm_gate<<<dim3(LSEQ), 256, 0, stream>>>(ycomb, zbuf, norm_w, y16);

    // out = y @ Wout
    gemm(y16, INNER, 0, WoutF16, DIMD, 0, outp, DIMD, 0, LSEQ, DIMD, INNER, 0, 1);

    (void)in_sizes; (void)n_in; (void)out_size; (void)ws_size;
}
